// _LongcatFlashMoE_85787676770798
// MI455X (gfx1250) — matrix operand order repaired and checked by reading, not compiled, not run on hardware
//
#include <hip/hip_runtime.h>

// ---------------------------------------------------------------------------
// LongCat Flash MoE, sparse top-2 routing, bf16 WMMA compute (gfx1250/CDNA5)
// B=2, S=2048 -> NT=4096 tokens, H=2048, INTER=1024, 8 routed + 4 zero experts
// GEMMs: 256 thr = 8 waves, block tile 64(M)x128(N), wave tile 32x32,
// double-buffered LDS fed by GLOBAL_LOAD_ASYNC_TO_LDS_B128 (ASYNCcnt).
// ---------------------------------------------------------------------------

typedef __bf16 bf16_t;
typedef __attribute__((ext_vector_type(16))) __bf16 v16bf;
typedef __attribute__((ext_vector_type(8)))  __bf16 v8bf;
typedef __attribute__((ext_vector_type(4)))  __bf16 v4bf;
typedef __attribute__((ext_vector_type(8)))  float  v8f;

union F16Frag { v16bf v; v8bf h[2]; };

#define NT     4096          // tokens (B*S)
#define Hdim   2048
#define INTER  1024
#define NE     8             // routed experts
#define NTOT   12            // routed + zero
#define SCALE_W 1.5f

static __device__ __forceinline__ bf16_t f2bf(float f) {
  unsigned u = __float_as_uint(f);
  unsigned r = u + 0x7FFFu + ((u >> 16) & 1u);
  unsigned short s = (unsigned short)(r >> 16);
  return __builtin_bit_cast(bf16_t, s);
}

static __device__ __forceinline__ void cvt4(const float* __restrict__ s,
                                            bf16_t* __restrict__ d) {
  float4 v = *(const float4*)s;
  v4bf o;
  o[0] = f2bf(v.x); o[1] = f2bf(v.y); o[2] = f2bf(v.z); o[3] = f2bf(v.w);
  *(v4bf*)d = o;
}

// CDNA5 async memory->LDS (16B per lane), tracked by ASYNCcnt.
static __device__ __forceinline__ void async_ld128(void* lds, const bf16_t* gaddr) {
  unsigned l = (unsigned)(unsigned long long)lds;   // low 32 bits = LDS offset
  asm volatile("global_load_async_to_lds_b128 %0, %1, off"
               :: "v"(l), "v"(gaddr) : "memory");
}
static __device__ __forceinline__ void wait_async0() {
  asm volatile("s_wait_asynccnt 0x0" ::: "memory");
}

static __device__ __forceinline__ v8f wmma_bf16(v16bf a, v16bf b, v8f c) {
  return __builtin_amdgcn_wmma_f32_16x16x32_bf16(false, a, false, b,
                                                 (short)0, c, false, false);
}

// ---------------------------------------------------------------------------
// K0: fp32 -> bf16 conversion of x and all expert weights; zero counters.
// ---------------------------------------------------------------------------
__global__ __launch_bounds__(256) void convert_kernel(
    const float* __restrict__ x,  const float* __restrict__ gw,
    const float* __restrict__ uw, const float* __restrict__ dw,
    bf16_t* __restrict__ xo, bf16_t* __restrict__ go,
    bf16_t* __restrict__ uo, bf16_t* __restrict__ dwo,
    int* __restrict__ cnt)
{
  size_t i = (size_t)blockIdx.x * blockDim.x + threadIdx.x;  // group of 4
  if (i < NE) cnt[i] = 0;
  size_t b = i * 4;
  if (i < (size_t)NT * Hdim / 4) cvt4(x + b, xo + b);
  cvt4(gw + b, go + b);
  cvt4(uw + b, uo + b);
  cvt4(dw + b, dwo + b);
}

// ---------------------------------------------------------------------------
// K1: router. One wave (32 lanes) per token. 8 tokens per 256-thread block.
// ---------------------------------------------------------------------------
__global__ __launch_bounds__(256) void router_kernel(
    const float* __restrict__ x, const float* __restrict__ cw,
    const float* __restrict__ bias,
    int* __restrict__ selE, float* __restrict__ selW, int* __restrict__ slotpos,
    float* __restrict__ zscale, int* __restrict__ cnt, int* __restrict__ tok_of)
{
  const int wave = threadIdx.x >> 5;
  const int lane = threadIdx.x & 31;
  const int t = blockIdx.x * 8 + wave;
  if (t >= NT) return;

  const float* xr = x + (size_t)t * Hdim;
  float xreg[Hdim / 32];
#pragma unroll
  for (int j = 0; j < Hdim / 32; ++j) xreg[j] = xr[j * 32 + lane];

  float logit[NTOT];
#pragma unroll
  for (int e = 0; e < NTOT; ++e) {
    const float* c = cw + (size_t)e * Hdim;
    float p = 0.f;
#pragma unroll
    for (int j = 0; j < Hdim / 32; ++j) p += xreg[j] * c[j * 32 + lane];
#pragma unroll
    for (int s = 16; s > 0; s >>= 1) p += __shfl_xor(p, s, 32);
    logit[e] = p;
  }

  float mx = logit[0];
#pragma unroll
  for (int e = 1; e < NTOT; ++e) mx = fmaxf(mx, logit[e]);
  float prob[NTOT], den = 0.f;
#pragma unroll
  for (int e = 0; e < NTOT; ++e) { prob[e] = __expf(logit[e] - mx); den += prob[e]; }
  const float inv = 1.f / den;
  float biased[NTOT];
#pragma unroll
  for (int e = 0; e < NTOT; ++e) { prob[e] *= inv; biased[e] = prob[e] + bias[e]; }

  int i0 = 0; float b0v = biased[0], p0 = prob[0];
#pragma unroll
  for (int e = 1; e < NTOT; ++e)
    if (biased[e] > b0v) { b0v = biased[e]; p0 = prob[e]; i0 = e; }
  int i1 = -1; float b1v = -3.0e38f, p1 = 0.f;
#pragma unroll
  for (int e = 0; e < NTOT; ++e)
    if (e != i0 && biased[e] > b1v) { b1v = biased[e]; p1 = prob[e]; i1 = e; }

  if (lane == 0) {
    float zs = 0.f;
    int   ids[2] = { i0, i1 };
    float ps[2]  = { p0, p1 };
#pragma unroll
    for (int k = 0; k < 2; ++k) {
      const int e = ids[k];
      const float w = ps[k] * SCALE_W;
      selE[t * 2 + k] = e;
      selW[t * 2 + k] = w;
      if (e < NE) {
        int p = atomicAdd(&cnt[e], 1);
        slotpos[t * 2 + k] = p;
        tok_of[e * NT + p] = t;
      } else {
        slotpos[t * 2 + k] = -1;
        zs += w;
      }
    }
    zscale[t] = zs;
  }
}

// K2: exclusive scan of 8 counts
__global__ void scan_kernel(const int* __restrict__ cnt, int* __restrict__ base) {
  if (threadIdx.x == 0) {
    int a = 0;
    for (int e = 0; e < NE; ++e) { base[e] = a; a += cnt[e]; }
  }
}

// K3: compacted gather list of token ids, grouped by expert
__global__ __launch_bounds__(256) void glist_kernel(
    const int* __restrict__ cnt, const int* __restrict__ base,
    const int* __restrict__ tok_of, int* __restrict__ glist)
{
  const int i = blockIdx.x * 256 + threadIdx.x;   // 8 * 4096
  const int e = i >> 12, p = i & (NT - 1);
  if (p < cnt[e]) glist[base[e] + p] = tok_of[i];
}

// ---------------------------------------------------------------------------
// K5: fused gate+up GEMM + SiLU.  M=cnt[e], K=2048, N=1024.
// Block tile 64x128, wave tile 32x32 (x2 matrices), K-step 32.
// ---------------------------------------------------------------------------
__global__ __launch_bounds__(256) void gateup_kernel(
    const bf16_t* __restrict__ xb, const bf16_t* __restrict__ gwb,
    const bf16_t* __restrict__ uwb,
    const int* __restrict__ glist, const int* __restrict__ cnt,
    const int* __restrict__ base, bf16_t* __restrict__ act)
{
  const int e = blockIdx.z;
  const int M = cnt[e];
  const int moff = blockIdx.y * 64;
  if (moff >= M) return;
  const int noff = blockIdx.x * 128;
  const int b0 = base[e];
  const bf16_t* G = gwb + (size_t)e * (INTER * Hdim);
  const bf16_t* U = uwb + (size_t)e * (INTER * Hdim);

  __shared__ bf16_t sA[2][64][32];     // 8 KB
  __shared__ bf16_t sG[2][128][32];    // 16 KB
  __shared__ bf16_t sU[2][128][32];    // 16 KB

  const int tid  = threadIdx.x;
  const int lane = tid & 31;
  const int wave = tid >> 5;
  const int wm = (wave >> 2) << 5;     // 0 / 32
  const int wn = (wave & 3) << 5;      // 0..96
  const int l16 = lane & 15;
  const bool hi = lane >= 16;

  // --- async staging assignments (16B per async op) ---
  // A: 64 rows x 32 cols = 256 chunks of 8 bf16 -> 1 per thread
  const int arow = tid >> 2, acol = (tid & 3) << 3;
  const int atok = glist[b0 + min(moff + arow, M - 1)];
  const bf16_t* aptr = xb + (size_t)atok * Hdim + acol;
  // B: 128 rows x 32 cols = 512 chunks -> 2 per thread (q = tid, tid+256)
  const int br0 = tid >> 2,          bc0 = (tid & 3) << 3;
  const int br1 = (tid + 256) >> 2,  bc1 = bc0;
  const bf16_t* gptr0 = G + (size_t)(noff + br0) * Hdim + bc0;
  const bf16_t* gptr1 = G + (size_t)(noff + br1) * Hdim + bc1;
  const bf16_t* uptr0 = U + (size_t)(noff + br0) * Hdim + bc0;
  const bf16_t* uptr1 = U + (size_t)(noff + br1) * Hdim + bc1;

  v8f cg[2][2] = {}; v8f cu[2][2] = {};

  // prologue: stage K-step 0 into buffer 0
  async_ld128(&sA[0][arow][acol], aptr);
  async_ld128(&sG[0][br0][bc0], gptr0);
  async_ld128(&sG[0][br1][bc1], gptr1);
  async_ld128(&sU[0][br0][bc0], uptr0);
  async_ld128(&sU[0][br1][bc1], uptr1);
  wait_async0();
  __syncthreads();

  const int KSTEPS = Hdim / 32;   // 64
  for (int ks = 0; ks < KSTEPS; ++ks) {
    const int cur = ks & 1;
    if (ks + 1 < KSTEPS) {
      const int nxt = cur ^ 1;
      const int ko = (ks + 1) * 32;
      async_ld128(&sA[nxt][arow][acol], aptr + ko);
      async_ld128(&sG[nxt][br0][bc0], gptr0 + ko);
      async_ld128(&sG[nxt][br1][bc1], gptr1 + ko);
      async_ld128(&sU[nxt][br0][bc0], uptr0 + ko);
      async_ld128(&sU[nxt][br1][bc1], uptr1 + ko);
      if (ks + 4 < KSTEPS) __builtin_prefetch(gptr0 + (ks + 4) * 32, 0, 3);
    }

    // fragments: A lane<16 -> K {0..15}; lane>=16 -> K {16..31}
    F16Frag a[2], bg[2], bu[2];
    const int ac = hi ? 16 : 0;
    const int bc = hi ? 16 : 0;
#pragma unroll
    for (int s = 0; s < 2; ++s) {
      const int ar = wm + s * 16 + l16;
      a[s].h[0] = *(const v8bf*)&sA[cur][ar][ac];
      a[s].h[1] = *(const v8bf*)&sA[cur][ar][ac + 8];
      const int bn = wn + s * 16 + l16;
      bg[s].v = *(const v16bf*)&sG[cur][bn][bc];
      bu[s].v = *(const v16bf*)&sU[cur][bn][bc];
    }
#pragma unroll
    for (int mi = 0; mi < 2; ++mi)
#pragma unroll
      for (int ni = 0; ni < 2; ++ni) {
        cg[mi][ni] = wmma_bf16(a[mi].v, bg[ni].v, cg[mi][ni]);
        cu[mi][ni] = wmma_bf16(a[mi].v, bu[ni].v, cu[mi][ni]);
      }

    wait_async0();
    __syncthreads();
  }

  // epilogue: C layout -> element (m = r + 8*hi, n = l16)
#pragma unroll
  for (int mi = 0; mi < 2; ++mi)
#pragma unroll
    for (int ni = 0; ni < 2; ++ni)
#pragma unroll
      for (int r = 0; r < 8; ++r) {
        const int m = moff + wm + mi * 16 + r + (hi ? 8 : 0);
        if (m < M) {
          const int n = noff + wn + ni * 16 + l16;
          const float g = cg[mi][ni][r], u = cu[mi][ni][r];
          const float s = g * __builtin_amdgcn_rcpf(1.f + __expf(-g));
          act[(size_t)(b0 + m) * INTER + n] = f2bf(s * u);
        }
      }
}

// ---------------------------------------------------------------------------
// K6: down GEMM. y[b0+m, n] = act[b0+m, :] . down[e, n, :].  K=1024, N=2048.
// ---------------------------------------------------------------------------
__global__ __launch_bounds__(256) void down_kernel(
    const bf16_t* __restrict__ act, const bf16_t* __restrict__ dwb,
    const int* __restrict__ cnt, const int* __restrict__ base,
    float* __restrict__ y)
{
  const int e = blockIdx.z;
  const int M = cnt[e];
  const int moff = blockIdx.y * 64;
  if (moff >= M) return;
  const int noff = blockIdx.x * 128;
  const int b0 = base[e];
  const bf16_t* D = dwb + (size_t)e * (Hdim * INTER);

  __shared__ bf16_t sA[2][64][32];
  __shared__ bf16_t sB[2][128][32];

  const int tid  = threadIdx.x;
  const int lane = tid & 31;
  const int wave = tid >> 5;
  const int wm = (wave >> 2) << 5;
  const int wn = (wave & 3) << 5;
  const int l16 = lane & 15;
  const bool hi = lane >= 16;

  const int arow = tid >> 2, acol = (tid & 3) << 3;
  const int am = min(moff + arow, M - 1);
  const bf16_t* aptr = act + (size_t)(b0 + am) * INTER + acol;
  const int br0 = tid >> 2,         bc0 = (tid & 3) << 3;
  const int br1 = (tid + 256) >> 2, bc1 = bc0;
  const bf16_t* bptr0 = D + (size_t)(noff + br0) * INTER + bc0;
  const bf16_t* bptr1 = D + (size_t)(noff + br1) * INTER + bc1;

  v8f c[2][2] = {};

  async_ld128(&sA[0][arow][acol], aptr);
  async_ld128(&sB[0][br0][bc0], bptr0);
  async_ld128(&sB[0][br1][bc1], bptr1);
  wait_async0();
  __syncthreads();

  const int KSTEPS = INTER / 32;  // 32
  for (int ks = 0; ks < KSTEPS; ++ks) {
    const int cur = ks & 1;
    if (ks + 1 < KSTEPS) {
      const int nxt = cur ^ 1;
      const int ko = (ks + 1) * 32;
      async_ld128(&sA[nxt][arow][acol], aptr + ko);
      async_ld128(&sB[nxt][br0][bc0], bptr0 + ko);
      async_ld128(&sB[nxt][br1][bc1], bptr1 + ko);
      if (ks + 4 < KSTEPS) __builtin_prefetch(bptr0 + (ks + 4) * 32, 0, 3);
    }

    F16Frag a[2], b[2];
    const int ac = hi ? 16 : 0;
    const int bc = hi ? 16 : 0;
#pragma unroll
    for (int s = 0; s < 2; ++s) {
      const int ar = wm + s * 16 + l16;
      a[s].h[0] = *(const v8bf*)&sA[cur][ar][ac];
      a[s].h[1] = *(const v8bf*)&sA[cur][ar][ac + 8];
      const int bn = wn + s * 16 + l16;
      b[s].v = *(const v16bf*)&sB[cur][bn][bc];
    }
#pragma unroll
    for (int mi = 0; mi < 2; ++mi)
#pragma unroll
      for (int ni = 0; ni < 2; ++ni)
        c[mi][ni] = wmma_bf16(a[mi].v, b[ni].v, c[mi][ni]);

    wait_async0();
    __syncthreads();
  }

#pragma unroll
  for (int mi = 0; mi < 2; ++mi)
#pragma unroll
    for (int ni = 0; ni < 2; ++ni)
#pragma unroll
      for (int r = 0; r < 8; ++r) {
        const int m = moff + wm + mi * 16 + r + (hi ? 8 : 0);
        if (m < M) {
          const int n = noff + wn + ni * 16 + l16;
          y[(size_t)(b0 + m) * Hdim + n] = c[mi][ni][r];
        }
      }
}

// ---------------------------------------------------------------------------
// K7: combine. out[t] = zscale*x[t] + w0*y[slot0] + w1*y[slot1] (fixed order).
// ---------------------------------------------------------------------------
__global__ __launch_bounds__(256) void combine_kernel(
    const float* __restrict__ x, const float* __restrict__ y,
    const int* __restrict__ selE, const float* __restrict__ selW,
    const int* __restrict__ slotpos, const float* __restrict__ zscale,
    const int* __restrict__ base, float* __restrict__ out)
{
  const int t = blockIdx.x;
  float w0 = 0.f, w1 = 0.f;
  size_t s0 = 0, s1 = 0;
  bool h0 = false, h1 = false;
  {
    int e = selE[t * 2 + 0];
    if (e < NE) { h0 = true; w0 = selW[t * 2 + 0];
                  s0 = (size_t)(base[e] + slotpos[t * 2 + 0]) * Hdim; }
    e = selE[t * 2 + 1];
    if (e < NE) { h1 = true; w1 = selW[t * 2 + 1];
                  s1 = (size_t)(base[e] + slotpos[t * 2 + 1]) * Hdim; }
  }
  const float zs = zscale[t];
  const float* xr = x + (size_t)t * Hdim;
  float* orow = out + (size_t)t * Hdim;

  for (int c = threadIdx.x * 4; c < Hdim; c += 256 * 4) {
    float4 xv = *(const float4*)(xr + c);
    float4 r;
    r.x = zs * xv.x; r.y = zs * xv.y; r.z = zs * xv.z; r.w = zs * xv.w;
    if (h0) {
      float4 a = *(const float4*)(y + s0 + c);
      r.x += w0 * a.x; r.y += w0 * a.y; r.z += w0 * a.z; r.w += w0 * a.w;
    }
    if (h1) {
      float4 a = *(const float4*)(y + s1 + c);
      r.x += w1 * a.x; r.y += w1 * a.y; r.z += w1 * a.z; r.w += w1 * a.w;
    }
    *(float4*)(orow + c) = r;
  }
}

// ---------------------------------------------------------------------------
// Launch
// ---------------------------------------------------------------------------
extern "C" void kernel_launch(void* const* d_in, const int* in_sizes, int n_in,
                              void* d_out, int out_size, void* d_ws, size_t ws_size,
                              hipStream_t stream) {
  const float* x    = (const float*)d_in[0];   // [2,2048,2048]
  const float* cw   = (const float*)d_in[1];   // [12,2048]
  const float* bias = (const float*)d_in[2];   // [12]
  const float* gw   = (const float*)d_in[3];   // [8,1024,2048]
  const float* uw   = (const float*)d_in[4];   // [8,1024,2048]
  const float* dw   = (const float*)d_in[5];   // [8,2048,1024]
  float* out = (float*)d_out;

  char* ws = (char*)d_ws;
  const size_t SZ_XB  = (size_t)NT * Hdim * 2;
  const size_t SZ_W   = (size_t)NE * INTER * Hdim * 2;
  const size_t SZ_ACT = (size_t)(2 * NT) * INTER * 2;
  const size_t SZ_Y   = (size_t)(2 * NT) * Hdim * 4;

  size_t o = 0;
  bf16_t* xb   = (bf16_t*)(ws + o); o += SZ_XB;
  bf16_t* gwb  = (bf16_t*)(ws + o); o += SZ_W;
  bf16_t* uwb  = (bf16_t*)(ws + o); o += SZ_W;
  bf16_t* dwb  = (bf16_t*)(ws + o); o += SZ_W;
  bf16_t* act  = (bf16_t*)(ws + o); o += SZ_ACT;
  float*  yb   = (float*)(ws + o);  o += SZ_Y;
  int*    selE = (int*)(ws + o);    o += (size_t)NT * 2 * 4;
  float*  selW = (float*)(ws + o);  o += (size_t)NT * 2 * 4;
  int*    slot = (int*)(ws + o);    o += (size_t)NT * 2 * 4;
  float*  zsc  = (float*)(ws + o);  o += (size_t)NT * 4;
  int*    cnt  = (int*)(ws + o);    o += 64;
  int*    base = (int*)(ws + o);    o += 64;
  int*    tokf = (int*)(ws + o);    o += (size_t)NE * NT * 4;
  int*    glist= (int*)(ws + o);    o += (size_t)(2 * NT) * 4;

  convert_kernel<<<16384, 256, 0, stream>>>(x, gw, uw, dw, xb, gwb, uwb, dwb, cnt);
  router_kernel<<<NT / 8, 256, 0, stream>>>(x, cw, bias, selE, selW, slot, zsc, cnt, tokf);
  scan_kernel<<<1, 32, 0, stream>>>(cnt, base);
  glist_kernel<<<(NE * NT) / 256, 256, 0, stream>>>(cnt, base, tokf, glist);
  gateup_kernel<<<dim3(INTER / 128, NT / 64, NE), 256, 0, stream>>>(
      xb, gwb, uwb, glist, cnt, base, act);
  down_kernel<<<dim3(Hdim / 128, NT / 64, NE), 256, 0, stream>>>(
      act, dwb, cnt, base, yb);
  combine_kernel<<<NT, 256, 0, stream>>>(x, yb, selE, selW, slot, zsc, base, out);
}
